// GCNV3_14448269984570
// MI455X (gfx1250) — matrix operand order repaired and checked by reading, not compiled, not run on hardware
//
#include <hip/hip_runtime.h>
#include <hip/hip_bf16.h>

#define N_NODES 40000
#define N_EDGES 640000
#define IN_DIM  512
#define HID_DIM 128
#define OUT_DIM 40

#define KSTEPS1 (IN_DIM / 32)    // 16 K-steps for GEMM1
#define NT1     (HID_DIM / 16)   // 8 N-tiles for GEMM1
#define KSTEPS2 (HID_DIM / 32)   // 4 K-steps for GEMM2
#define NT2     3                // 48-col padded N for GEMM2

typedef __attribute__((ext_vector_type(16))) __bf16    v16bf;
typedef __attribute__((ext_vector_type(8)))  float     v8f;
typedef __attribute__((ext_vector_type(8)))  unsigned  v8u;

// pack two f32 into one dword of two bf16 (round-to-nearest-even), pure int ops
static __device__ __forceinline__ unsigned pk_bf16(float lo, float hi) {
    unsigned ul = __builtin_bit_cast(unsigned, lo);
    unsigned uh = __builtin_bit_cast(unsigned, hi);
    ul = (ul + 0x7FFFu + ((ul >> 16) & 1u)) >> 16;
    uh = (uh + 0x7FFFu + ((uh >> 16) & 1u)) & 0xFFFF0000u;
    return ul | uh;
}

__global__ void zero_kernel(float* __restrict__ p, int n) {
    int i = blockIdx.x * blockDim.x + threadIdx.x;
    if (i < n) p[i] = 0.0f;
}

// ---------------------------------------------------------------------------
// Pre-pack W1[512,128] into bf16 B-fragments, WMMA lane order:
// frag = t*KSTEPS1 + ks ; lane L holds col t*16+(L&15), K = ks*32+(L>>4)*8 + j + (j>=8)*8
// Each lane's 16 bf16 (32B) stored contiguously -> GEMM loads one v16bf/lane.
// ---------------------------------------------------------------------------
__global__ void pack_w1_kernel(const float* __restrict__ W1, __bf16* __restrict__ W1bf) {
    const int tid = blockIdx.x * 256 + threadIdx.x;       // NT1*KSTEPS1*32 = 4096
    if (tid >= NT1 * KSTEPS1 * 32) return;
    const int lane  = tid & 31;
    const int frag  = tid >> 5;
    const int ks    = frag % KSTEPS1;
    const int t     = frag / KSTEPS1;
    const int n     = t * 16 + (lane & 15);
    const int kbase = ks * 32 + (lane >> 4) * 8;
    unsigned* out = (unsigned*)W1bf + (size_t)tid * 8;
    #pragma unroll
    for (int p = 0; p < 8; ++p) {
        const float lo = W1[(size_t)(kbase + 2 * p + (p >> 2) * 8)     * HID_DIM + n];
        const float hi = W1[(size_t)(kbase + 2 * p + (p >> 2) * 8 + 1) * HID_DIM + n];
        out[p] = pk_bf16(lo, hi);
    }
}

// Same for W2[128,40], padded to 48 cols with zeros.
__global__ void pack_w2_kernel(const float* __restrict__ W2, __bf16* __restrict__ W2bf) {
    const int tid = blockIdx.x * 256 + threadIdx.x;       // NT2*KSTEPS2*32 = 384
    if (tid >= NT2 * KSTEPS2 * 32) return;
    const int lane  = tid & 31;
    const int frag  = tid >> 5;
    const int ks    = frag % KSTEPS2;
    const int t     = frag / KSTEPS2;
    const int n     = t * 16 + (lane & 15);
    const int kbase = ks * 32 + (lane >> 4) * 8;
    unsigned* out = (unsigned*)W2bf + (size_t)tid * 8;
    #pragma unroll
    for (int p = 0; p < 8; ++p) {
        const float lo = (n < OUT_DIM) ? W2[(size_t)(kbase + 2 * p + (p >> 2) * 8)     * OUT_DIM + n] : 0.0f;
        const float hi = (n < OUT_DIM) ? W2[(size_t)(kbase + 2 * p + (p >> 2) * 8 + 1) * OUT_DIM + n] : 0.0f;
        out[p] = pk_bf16(lo, hi);
    }
}

// Build A fragment (16-bit A 16x32 layout) from 2x 32B contiguous f32 chunks.
static __device__ __forceinline__ v16bf load_a_frag(const float* __restrict__ ap) {
    const float4 c0 = *(const float4*)(ap);
    const float4 c1 = *(const float4*)(ap + 4);
    const float4 c2 = *(const float4*)(ap + 16);
    const float4 c3 = *(const float4*)(ap + 20);
    v8u au;
    au[0] = pk_bf16(c0.x, c0.y); au[1] = pk_bf16(c0.z, c0.w);
    au[2] = pk_bf16(c1.x, c1.y); au[3] = pk_bf16(c1.z, c1.w);
    au[4] = pk_bf16(c2.x, c2.y); au[5] = pk_bf16(c2.z, c2.w);
    au[6] = pk_bf16(c3.x, c3.y); au[7] = pk_bf16(c3.z, c3.w);
    return __builtin_bit_cast(v16bf, au);
}

// ---------------------------------------------------------------------------
// GEMM1: y1 = x @ W1. Block = 128 threads (4 waves); wave owns a 16-row stripe
// and all 8 N-tiles. All 8 B fragments of a K-step are preloaded (one load
// clause), then the 8 WMMAs issue back-to-back behind a single wait.
// ---------------------------------------------------------------------------
__global__ __launch_bounds__(128) void gemm1_kernel(const float* __restrict__ x,
                                                    const __bf16* __restrict__ W1bf,
                                                    float* __restrict__ y1) {
    const int lane = threadIdx.x & 31;
    const int wave = threadIdx.x >> 5;
    const int hiw  = lane >> 4;
    const int mr   = lane & 15;
    const int m0   = blockIdx.x * 64 + wave * 16;
    const int m    = m0 + mr;

    v8f acc[NT1] = {};

    for (int ks = 0; ks < KSTEPS1; ++ks) {
        // batch-load all B fragments for this K-step
        v16bf bfrag[NT1];
        #pragma unroll
        for (int t = 0; t < NT1; ++t)
            bfrag[t] = *(const v16bf*)(W1bf +
                ((size_t)(t * KSTEPS1 + ks) * 32 + lane) * 16);
        const v16bf a = load_a_frag(x + (size_t)m * IN_DIM + ks * 32 + hiw * 8);
        #pragma unroll
        for (int t = 0; t < NT1; ++t)
            acc[t] = __builtin_amdgcn_wmma_f32_16x16x32_bf16(
                false, a, false, bfrag[t], (short)0, acc[t], false, false);
    }
    #pragma unroll
    for (int t = 0; t < NT1; ++t) {
        const int n = t * 16 + mr;
        #pragma unroll
        for (int r = 0; r < 8; ++r)
            y1[(size_t)(m0 + r + 8 * hiw) * HID_DIM + n] = acc[t][r];
    }
}

// ---------------------------------------------------------------------------
// SpMM1: hacc[row] += val * y1[col], 128 dims. One wave per edge, float4/lane.
// ---------------------------------------------------------------------------
__global__ __launch_bounds__(256) void spmm1_kernel(const int* __restrict__ erow,
                                                    const int* __restrict__ ecol,
                                                    const float* __restrict__ eval_,
                                                    const float* __restrict__ y1,
                                                    float* __restrict__ hacc) {
    const int tid  = blockIdx.x * 256 + threadIdx.x;
    const int e    = tid >> 5;
    const int lane = tid & 31;
    if (e >= N_EDGES) return;
    const int   r = erow[e];
    const int   c = ecol[e];
    const float v = eval_[e];
    const float4 mseg = ((const float4*)(y1 + (size_t)c * HID_DIM))[lane];
    float* dst = hacc + (size_t)r * HID_DIM + lane * 4;
    atomicAdd(dst + 0, v * mseg.x);
    atomicAdd(dst + 1, v * mseg.y);
    atomicAdd(dst + 2, v * mseg.z);
    atomicAdd(dst + 3, v * mseg.w);
}

// ---------------------------------------------------------------------------
// GEMM2: y2 = relu(hacc + b1) @ W2 (N padded to 48; cols >= 40 are zero in
// the pre-packed B). Bias+ReLU fused into the A conversion.
// ---------------------------------------------------------------------------
__global__ __launch_bounds__(128) void gemm2_kernel(const float* __restrict__ hacc,
                                                    const float* __restrict__ b1,
                                                    const __bf16* __restrict__ W2bf,
                                                    float* __restrict__ y2) {
    const int lane = threadIdx.x & 31;
    const int wave = threadIdx.x >> 5;
    const int hiw  = lane >> 4;
    const int mr   = lane & 15;
    const int m0   = blockIdx.x * 64 + wave * 16;
    const int m    = m0 + mr;

    v8f acc[NT2] = {};

    #pragma unroll
    for (int ks = 0; ks < KSTEPS2; ++ks) {
        v16bf bfrag[NT2];
        #pragma unroll
        for (int t = 0; t < NT2; ++t)
            bfrag[t] = *(const v16bf*)(W2bf +
                ((size_t)(t * KSTEPS2 + ks) * 32 + lane) * 16);

        const float* ap = hacc + (size_t)m * HID_DIM + ks * 32 + hiw * 8;
        const float* bb = b1 + ks * 32 + hiw * 8;
        const float4 c0 = *(const float4*)(ap);
        const float4 c1 = *(const float4*)(ap + 4);
        const float4 c2 = *(const float4*)(ap + 16);
        const float4 c3 = *(const float4*)(ap + 20);
        const float4 d0 = *(const float4*)(bb);
        const float4 d1 = *(const float4*)(bb + 4);
        const float4 d2 = *(const float4*)(bb + 16);
        const float4 d3 = *(const float4*)(bb + 20);
        v8u au;
        au[0] = pk_bf16(fmaxf(c0.x + d0.x, 0.f), fmaxf(c0.y + d0.y, 0.f));
        au[1] = pk_bf16(fmaxf(c0.z + d0.z, 0.f), fmaxf(c0.w + d0.w, 0.f));
        au[2] = pk_bf16(fmaxf(c1.x + d1.x, 0.f), fmaxf(c1.y + d1.y, 0.f));
        au[3] = pk_bf16(fmaxf(c1.z + d1.z, 0.f), fmaxf(c1.w + d1.w, 0.f));
        au[4] = pk_bf16(fmaxf(c2.x + d2.x, 0.f), fmaxf(c2.y + d2.y, 0.f));
        au[5] = pk_bf16(fmaxf(c2.z + d2.z, 0.f), fmaxf(c2.w + d2.w, 0.f));
        au[6] = pk_bf16(fmaxf(c3.x + d3.x, 0.f), fmaxf(c3.y + d3.y, 0.f));
        au[7] = pk_bf16(fmaxf(c3.z + d3.z, 0.f), fmaxf(c3.w + d3.w, 0.f));
        const v16bf a = __builtin_bit_cast(v16bf, au);
        #pragma unroll
        for (int t = 0; t < NT2; ++t)
            acc[t] = __builtin_amdgcn_wmma_f32_16x16x32_bf16(
                false, a, false, bfrag[t], (short)0, acc[t], false, false);
    }
    #pragma unroll
    for (int t = 0; t < NT2; ++t) {
        const int n = t * 16 + mr;
        if (n < OUT_DIM) {
            #pragma unroll
            for (int r = 0; r < 8; ++r)
                y2[(size_t)(m0 + r + 8 * hiw) * OUT_DIM + n] = acc[t][r];
        }
    }
}

// ---------------------------------------------------------------------------
// SpMM2: out[row] += val * y2[col], 40 dims. One thread per (edge, dim).
// ---------------------------------------------------------------------------
__global__ __launch_bounds__(256) void spmm2_kernel(const int* __restrict__ erow,
                                                    const int* __restrict__ ecol,
                                                    const float* __restrict__ eval_,
                                                    const float* __restrict__ y2,
                                                    float* __restrict__ oacc) {
    const int tid = blockIdx.x * 256 + threadIdx.x;
    if (tid >= N_EDGES * OUT_DIM) return;
    const int e = tid / OUT_DIM;
    const int d = tid - e * OUT_DIM;
    atomicAdd(oacc + (size_t)erow[e] * OUT_DIM + d,
              eval_[e] * y2[(size_t)ecol[e] * OUT_DIM + d]);
}

// ---------------------------------------------------------------------------
// log_softmax over 40 columns, in place on d_out, bias b2 fused.
// ---------------------------------------------------------------------------
__global__ __launch_bounds__(256) void logsoftmax_kernel(const float* __restrict__ b2,
                                                         float* __restrict__ out) {
    const int node = blockIdx.x * 256 + threadIdx.x;
    if (node >= N_NODES) return;
    float* row = out + (size_t)node * OUT_DIM;
    float v[OUT_DIM];
    float mx = -3.402823466e+38f;
    #pragma unroll
    for (int d = 0; d < OUT_DIM; ++d) {
        v[d] = row[d] + b2[d];
        mx = fmaxf(mx, v[d]);
    }
    float s = 0.0f;
    #pragma unroll
    for (int d = 0; d < OUT_DIM; ++d) s += __expf(v[d] - mx);
    const float lse = mx + __logf(s);
    #pragma unroll
    for (int d = 0; d < OUT_DIM; ++d) row[d] = v[d] - lse;
}

extern "C" void kernel_launch(void* const* d_in, const int* in_sizes, int n_in,
                              void* d_out, int out_size, void* d_ws, size_t ws_size,
                              hipStream_t stream) {
    const float* x     = (const float*)d_in[0];
    const int*   erow  = (const int*)  d_in[1];
    const int*   ecol  = (const int*)  d_in[2];
    const float* eval_ = (const float*)d_in[3];
    const float* W1    = (const float*)d_in[4];
    const float* b1    = (const float*)d_in[5];
    const float* W2    = (const float*)d_in[6];
    const float* b2    = (const float*)d_in[7];
    float* out = (float*)d_out;

    float*  y1   = (float*)d_ws;                        // 40000*128 f32
    float*  hacc = y1   + (size_t)N_NODES * HID_DIM;    // 40000*128 f32
    float*  y2   = hacc + (size_t)N_NODES * HID_DIM;    // 40000*40  f32
    __bf16* W1bf = (__bf16*)(y2 + (size_t)N_NODES * OUT_DIM);   // 65536 bf16
    __bf16* W2bf = W1bf + (size_t)NT1 * KSTEPS1 * 32 * 16;      // 6144 bf16

    zero_kernel<<<(N_NODES * HID_DIM + 255) / 256, 256, 0, stream>>>(hacc, N_NODES * HID_DIM);
    zero_kernel<<<(N_NODES * OUT_DIM + 255) / 256, 256, 0, stream>>>(out, N_NODES * OUT_DIM);
    pack_w1_kernel<<<(NT1 * KSTEPS1 * 32 + 255) / 256, 256, 0, stream>>>(W1, W1bf);
    pack_w2_kernel<<<(NT2 * KSTEPS2 * 32 + 255) / 256, 256, 0, stream>>>(W2, W2bf);

    gemm1_kernel<<<N_NODES / 64, 128, 0, stream>>>(x, W1bf, y1);
    spmm1_kernel<<<(N_EDGES * 32) / 256, 256, 0, stream>>>(erow, ecol, eval_, y1, hacc);
    gemm2_kernel<<<N_NODES / 64, 128, 0, stream>>>(hacc, b1, W2bf, y2);
    spmm2_kernel<<<(N_EDGES * OUT_DIM + 255) / 256, 256, 0, stream>>>(erow, ecol, eval_, y2, out);
    logsoftmax_kernel<<<(N_NODES + 255) / 256, 256, 0, stream>>>(b2, out);
}
